// PlusPlusNetwork_17145509446338
// MI455X (gfx1250) — matrix operand order repaired and checked by reading, not compiled, not run on hardware
//
#include <hip/hip_runtime.h>
#include <hip/hip_bf16.h>
#include <math.h>

// ---------------------------------------------------------------------------
// Types for CDNA5 WMMA (wave32): 16x16x32 bf16 -> f32
// ---------------------------------------------------------------------------
typedef __attribute__((ext_vector_type(16))) __bf16 v16bf;
typedef __attribute__((ext_vector_type(8)))  __bf16 v8bf;
typedef __attribute__((ext_vector_type(8)))  float  v8f;
typedef __attribute__((__vector_size__(16))) int    v4i;  // matches builtin proto

#define EPSF 1e-15f
#define MAXN 0.996f   /* 1 - 4e-3 (geoopt boundary eps) */

#if __has_builtin(__builtin_amdgcn_global_load_async_to_lds_b128)
#define HAVE_ASYNC_LDS 1
#else
#define HAVE_ASYNC_LDS 0
#endif

// 16-byte global -> LDS copy. Async (ASYNCcnt-tracked, no VGPR round trip)
// when the toolchain exposes the gfx1250 builtin; sync fallback otherwise.
__device__ __forceinline__ void copy_b128(__bf16* lds, const __bf16* g) {
#if HAVE_ASYNC_LDS
  __builtin_amdgcn_global_load_async_to_lds_b128(
      (__attribute__((address_space(1))) v4i*)g,
      (__attribute__((address_space(3))) v4i*)lds, /*offset=*/0, /*cpol=*/0);
#else
  *(uint4*)lds = *(const uint4*)g;
#endif
}

__device__ __forceinline__ void wait_async() {
#if HAVE_ASYNC_LDS
#if __has_builtin(__builtin_amdgcn_s_wait_asynccnt)
  __builtin_amdgcn_s_wait_asynccnt(0);
#else
  asm volatile("s_wait_asynccnt 0x0" ::: "memory");
#endif
#endif
}

__device__ __forceinline__ float half_wave_reduce16(float v) {
  // reduce across the 16-lane half of a wave32 (xor masks stay inside group)
  v += __shfl_xor(v, 1, 32);
  v += __shfl_xor(v, 2, 32);
  v += __shfl_xor(v, 4, 32);
  v += __shfl_xor(v, 8, 32);
  return v;
}

// ---------------------------------------------------------------------------
// h0 = project(expmap0(x)) in bf16, plus cx2[row] = ||h0||^2
// one block per row (D = 512)
// ---------------------------------------------------------------------------
__global__ void expmap0_kernel(const float* __restrict__ x,
                               __bf16* __restrict__ h,
                               float* __restrict__ cx2, int D) {
  __shared__ float red[256];
  const int row = blockIdx.x;
  const float* xr = x + (size_t)row * D;
  float s = 0.f;
  for (int c = threadIdx.x; c < D; c += 256) { float v = xr[c]; s += v * v; }
  red[threadIdx.x] = s;
  __syncthreads();
  for (int off = 128; off > 0; off >>= 1) {
    if (threadIdx.x < off) red[threadIdx.x] += red[threadIdx.x + off];
    __syncthreads();
  }
  float n  = fmaxf(sqrtf(red[0]), EPSF);
  float th = tanhf(n);
  float scale = th / n;
  if (th > MAXN) scale = MAXN / n;       // boundary projection
  float hn = fminf(th, MAXN);
  if (threadIdx.x == 0) cx2[row] = hn * hn;
  __bf16* hr = h + (size_t)row * D;
  for (int c = threadIdx.x; c < D; c += 256) hr[c] = (__bf16)(xr[c] * scale);
}

// ---------------------------------------------------------------------------
// Weight prep: wT[n][k] = bf16( wv[k][n] / ||wv[:,n]|| )  (z_unit transposed)
// one block per output column n
// ---------------------------------------------------------------------------
__global__ void wprep_kernel(const float* __restrict__ wv,
                             __bf16* __restrict__ wT, int K, int N) {
  __shared__ float red[256];
  const int n = blockIdx.x;
  float s = 0.f;
  for (int k = threadIdx.x; k < K; k += 256) {
    float v = wv[(size_t)k * N + n];
    s += v * v;
  }
  red[threadIdx.x] = s;
  __syncthreads();
  for (int off = 128; off > 0; off >>= 1) {
    if (threadIdx.x < off) red[threadIdx.x] += red[threadIdx.x + off];
    __syncthreads();
  }
  float inv = 1.f / fmaxf(sqrtf(red[0]), EPSF);
  for (int k = threadIdx.x; k < K; k += 256)
    wT[(size_t)n * K + k] = (__bf16)(wv[(size_t)k * N + n] * inv);
}

__global__ void zero_kernel(float* __restrict__ p, int n) {
  int i = blockIdx.x * 256 + threadIdx.x;
  if (i < n) p[i] = 0.f;
}

// ---------------------------------------------------------------------------
// Fused MLR GEMM: acc = A(bf16[M,K]) @ z_unit  via  Bt = z_unit^T (bf16[N,K])
//   num = 2*acc*cosh(2r) - (1+cx2)*sinh(2r);  den = max(1-cx2, eps)
//   non-head: y = sinh(2g*asinh(num/den)) -> Y bf16, rowsum += y^2 (atomic)
//   head:     out = 2g*asinh(num/den)     -> f32
// Block: 256 thr = 8 waves; tile 128x128; wave tile 32(M) x 64(N).
// Double-buffered LDS; async global->LDS copies overlap with WMMA compute.
// ---------------------------------------------------------------------------
template <bool HEAD>
__global__ __launch_bounds__(256) void mlr_gemm_kernel(
    const __bf16* __restrict__ A, const __bf16* __restrict__ Bt,
    const float* __restrict__ cx2, const float* __restrict__ g,
    const float* __restrict__ r, __bf16* __restrict__ Y,
    float* __restrict__ rowsum, float* __restrict__ Out, int K, int N) {
  __shared__ __bf16 As[2][128 * 72];  // 128 rows x 64 bf16, +8 pad (bank spread)
  __shared__ __bf16 Bs[2][128 * 72];

  const int tid   = threadIdx.x;
  const int wave  = tid >> 5;
  const int lane  = tid & 31;
  const int hi    = lane >> 4;   // half-wave select
  const int nlane = lane & 15;
  const int waveM = wave >> 1;   // 0..3
  const int waveN = wave & 1;    // 0..1
  const int rowBase = blockIdx.y * 128;
  const int colBase = blockIdx.x * 128;

  v8f acc[2][4];
#pragma unroll
  for (int mb = 0; mb < 2; ++mb)
#pragma unroll
    for (int nb = 0; nb < 4; ++nb)
#pragma unroll
      for (int e = 0; e < 8; ++e) acc[mb][nb][e] = 0.f;

  const int nKc = K >> 6;  // K chunks of 64

  // stage one 128x64 tile pair (A,Bt) into LDS buffer `buf`
  auto stage = [&](int buf, int kc) {
    const int k0 = kc << 6;
#pragma unroll
    for (int j = 0; j < 4; ++j) {
      int idx = tid + j * 256;  // 1024 vec8 slots per tile
      int rr = idx >> 3, vv = idx & 7;
      copy_b128(&As[buf][rr * 72 + vv * 8],
                A + (size_t)(rowBase + rr) * K + k0 + vv * 8);
      copy_b128(&Bs[buf][rr * 72 + vv * 8],
                Bt + (size_t)(colBase + rr) * K + k0 + vv * 8);
    }
  };

  stage(0, 0);
  wait_async();
  __syncthreads();

  for (int kc = 0; kc < nKc; ++kc) {
    const int buf = kc & 1;
    if (kc + 1 < nKc) stage(buf ^ 1, kc + 1);  // overlap with compute below
    const __bf16* Asb = As[buf];
    const __bf16* Bsb = Bs[buf];

#pragma unroll
    for (int kk = 0; kk < 64; kk += 32) {
      // A frag (16x32 bf16, ISA 7.12.2): lanes0-15 K=[0..7,16..23], lanes16-31 +8
      v16bf afrag[2];
#pragma unroll
      for (int mb = 0; mb < 2; ++mb) {
        const __bf16* p = &Asb[(waveM * 32 + mb * 16 + nlane) * 72 + kk + hi * 8];
        v8bf lo = *(const v8bf*)p;
        v8bf hh = *(const v8bf*)(p + 16);
#pragma unroll
        for (int e = 0; e < 8; ++e) { afrag[mb][e] = lo[e]; afrag[mb][e + 8] = hh[e]; }
      }
      // B frag (32x16 bf16): lane = column, lanes0-15 K=[0..7,16..23], lanes16-31 +8
      v16bf bfrag[4];
#pragma unroll
      for (int nb = 0; nb < 4; ++nb) {
        const __bf16* p = &Bsb[(waveN * 64 + nb * 16 + nlane) * 72 + kk + hi * 8];
        v8bf lo = *(const v8bf*)p;
        v8bf hh = *(const v8bf*)(p + 16);
#pragma unroll
        for (int e = 0; e < 8; ++e) { bfrag[nb][e] = lo[e]; bfrag[nb][e + 8] = hh[e]; }
      }
#pragma unroll
      for (int mb = 0; mb < 2; ++mb)
#pragma unroll
        for (int nb = 0; nb < 4; ++nb)
          acc[mb][nb] = __builtin_amdgcn_wmma_f32_16x16x32_bf16(
              false, afrag[mb], false, bfrag[nb], (short)0, acc[mb][nb],
              false, false);
    }
    // next buffer's async copies done (they overlapped the WMMAs above);
    // barrier also guarantees everyone finished reading buf before re-staging
    wait_async();
    __syncthreads();
  }

  // ----- fused MLR epilogue -----
  float chv[4], shv[4], g2v[4];
  int colg[4];
#pragma unroll
  for (int nb = 0; nb < 4; ++nb) {
    int col = colBase + waveN * 64 + nb * 16 + nlane;
    colg[nb] = col;
    float d = 2.f * r[col];
    chv[nb] = coshf(d);
    shv[nb] = sinhf(d);
    g2v[nb] = 2.f * g[col];
  }
#pragma unroll
  for (int mb = 0; mb < 2; ++mb) {
#pragma unroll
    for (int i = 0; i < 8; ++i) {
      int row = rowBase + waveM * 32 + mb * 16 + hi * 8 + i;  // C layout 7.12.2
      float c2   = cx2[row];
      float den  = fmaxf(1.f - c2, EPSF);
      float onep = 1.f + c2;
      float s2 = 0.f;
#pragma unroll
      for (int nb = 0; nb < 4; ++nb) {
        float dot = acc[mb][nb][i];
        float num = 2.f * dot * chv[nb] - onep * shv[nb];
        float mlr = g2v[nb] * asinhf(num / den);
        if (HEAD) {
          Out[(size_t)row * N + colg[nb]] = mlr;
        } else {
          float y = sinhf(mlr);
          Y[(size_t)row * N + colg[nb]] = (__bf16)y;
          s2 += y * y;
        }
      }
      if (!HEAD) {
        s2 = half_wave_reduce16(s2);
        if (nlane == 0) atomicAdd(&rowsum[row], s2);
      }
    }
  }
}

// ---------------------------------------------------------------------------
// finalize: h = project(y / (1 + sqrt(1 + sum y^2))) in place; emit next cx2
// one block per row
// ---------------------------------------------------------------------------
__global__ void finalize_kernel(__bf16* __restrict__ h,
                                const float* __restrict__ rowsum,
                                float* __restrict__ cx2, int N) {
  const int row = blockIdx.x;
  float S = rowsum[row];
  float denom = 1.f + sqrtf(1.f + S);
  float inv = 1.f / denom;
  float nrm2 = S * inv * inv;
  float nrm = sqrtf(nrm2);
  float scale = inv;
  if (nrm > MAXN) scale = MAXN / (denom * nrm);
  if (threadIdx.x == 0) { float m = fminf(nrm, MAXN); cx2[row] = m * m; }
  __bf16* hr = h + (size_t)row * N;
  for (int c = threadIdx.x; c < N; c += 256)
    hr[c] = (__bf16)((float)hr[c] * scale);
}

// ---------------------------------------------------------------------------
extern "C" void kernel_launch(void* const* d_in, const int* in_sizes, int n_in,
                              void* d_out, int out_size, void* d_ws,
                              size_t ws_size, hipStream_t stream) {
  const float* x   = (const float*)d_in[0];
  const float* wg0 = (const float*)d_in[1];
  const float* wv0 = (const float*)d_in[2];
  const float* b0  = (const float*)d_in[3];
  const float* wg1 = (const float*)d_in[4];
  const float* wv1 = (const float*)d_in[5];
  const float* b1  = (const float*)d_in[6];
  const float* wg2 = (const float*)d_in[7];
  const float* wv2 = (const float*)d_in[8];
  const float* b2  = (const float*)d_in[9];
  const float* hg  = (const float*)d_in[10];
  const float* hv  = (const float*)d_in[11];
  const float* hb  = (const float*)d_in[12];
  float* out = (float*)d_out;

  const int B = 16384;
  const size_t HBYTES = (size_t)B * 2048 * 2;  // 64 MiB per ping-pong buffer
  char* ws = (char*)d_ws;
  __bf16* hA     = (__bf16*)ws;
  __bf16* hB     = (__bf16*)(ws + HBYTES);
  __bf16* wT     = (__bf16*)(ws + 2 * HBYTES);            // up to 2048x2048 bf16
  float*  rowsum = (float*)(ws + 2 * HBYTES + 8388608);
  float*  cx2    = rowsum + B;

  // stage 0: expmap0 -> hA bf16 [B,512], cx2
  expmap0_kernel<<<B, 256, 0, stream>>>(x, hA, cx2, 512);

  // layer 0: 512 -> 2048  (hA -> hB)
  wprep_kernel<<<2048, 256, 0, stream>>>(wv0, wT, 512, 2048);
  zero_kernel<<<64, 256, 0, stream>>>(rowsum, B);
  mlr_gemm_kernel<false><<<dim3(2048 / 128, B / 128), 256, 0, stream>>>(
      hA, wT, cx2, wg0, b0, hB, rowsum, nullptr, 512, 2048);
  finalize_kernel<<<B, 256, 0, stream>>>(hB, rowsum, cx2, 2048);

  // layer 1: 2048 -> 1024  (hB -> hA)
  wprep_kernel<<<1024, 256, 0, stream>>>(wv1, wT, 2048, 1024);
  zero_kernel<<<64, 256, 0, stream>>>(rowsum, B);
  mlr_gemm_kernel<false><<<dim3(1024 / 128, B / 128), 256, 0, stream>>>(
      hB, wT, cx2, wg1, b1, hA, rowsum, nullptr, 2048, 1024);
  finalize_kernel<<<B, 256, 0, stream>>>(hA, rowsum, cx2, 1024);

  // layer 2: 1024 -> 512  (hA -> hB)
  wprep_kernel<<<512, 256, 0, stream>>>(wv2, wT, 1024, 512);
  zero_kernel<<<64, 256, 0, stream>>>(rowsum, B);
  mlr_gemm_kernel<false><<<dim3(512 / 128, B / 128), 256, 0, stream>>>(
      hA, wT, cx2, wg2, b2, hB, rowsum, nullptr, 1024, 512);
  finalize_kernel<<<B, 256, 0, stream>>>(hB, rowsum, cx2, 512);

  // head: 512 -> 256  (hB -> out f32)
  wprep_kernel<<<256, 256, 0, stream>>>(hv, wT, 512, 256);
  mlr_gemm_kernel<true><<<dim3(256 / 128, B / 128), 256, 0, stream>>>(
      hB, wT, cx2, hg, hb, nullptr, nullptr, out, 512, 256);
}
